// LeNet_5_54546084659361
// MI455X (gfx1250) — matrix operand order repaired and checked by reading, compiled, not run on hardware
//
#include <hip/hip_runtime.h>
#include <hip/hip_bf16.h>
#include <math.h>

// ---------------- CDNA5 WMMA types ----------------
typedef __attribute__((ext_vector_type(16))) _Float16 v16h;
typedef __attribute__((ext_vector_type(8)))  float    v8f;
typedef __attribute__((ext_vector_type(4)))  int      v4i;

union Frag {
    v16h     h;
    v4i      i[2];
    int      w[8];
    _Float16 e[16];
};

__device__ __forceinline__ v8f wmma_f16(const Frag& a, const Frag& b, v8f c) {
    return __builtin_amdgcn_wmma_f32_16x16x32_f16(
        /*neg_a=*/false, a.h, /*neg_b=*/false, b.h,
        /*c_mod=*/(short)0, c, /*reuse_a=*/false, /*reuse_b=*/false);
}

// Branch-free tanh on the hardware transcendentals:
//   tanh(x) = 1 - 2/(1 + e^{2x}),  e^{2x} = 2^{x * 2*log2(e)}
// Saturates correctly: exp2->inf => +1 ; exp2->0 => -1.
__device__ __forceinline__ float fast_tanh(float x) {
    float e = __builtin_amdgcn_exp2f(x * 2.885390081777927f);
    return 1.f - 2.f * __builtin_amdgcn_rcpf(1.f + e);
}

// C3 connection mask (bitmask of S2 channels feeding each C3 map)
__constant__ unsigned char kC3Mask[16] = {
    0x07, 0x0E, 0x1C, 0x38, 0x31, 0x23,
    0x0F, 0x1E, 0x3C, 0x39, 0x33, 0x27,
    0x1B, 0x36, 0x2D, 0x3F
};

// ---------------------------------------------------------------
// Kernel 0: pack/convert all GEMM weights to f16, [N][K] row-major,
// zero-padded so every WMMA K-step / N-tile reads valid data.
//   w1pe: [16][64] C1 even-column taps, K = b*8 + t, b=ky(<5), t=kx(<5)
//   w1po: [16][64] C1 odd-column taps shifted one slot (t=1..5 real)
//         -> odd conv columns reuse the SAME even-aligned A run.
//   w3p : [16][160] C3, mask applied, K = ky*32 + (kx*6 + c)
//   w5p : [128][416] C5  (K = ch*25 + tap, matches s4 flatten)
//   l1p : [96][128] F6
//   l2p : [16][96]  output
// ---------------------------------------------------------------
__global__ void lenet_prep(const float* __restrict__ c1w,
                           const float* __restrict__ c3w,
                           const float* __restrict__ c5w,
                           const float* __restrict__ l1w,
                           const float* __restrict__ l2w,
                           _Float16* __restrict__ w1pe,
                           _Float16* __restrict__ w1po,
                           _Float16* __restrict__ w3p,
                           _Float16* __restrict__ w5p,
                           _Float16* __restrict__ l1p,
                           _Float16* __restrict__ l2p) {
    const int N3 = 16 * 160;
    const int N5 = 128 * 416;
    const int N1 = 96 * 128;
    const int N2 = 16 * 96;
    const int NC = 16 * 64;
    int i = blockIdx.x * blockDim.x + threadIdx.x;
    if (i < N3) {
        int n = i / 160, k = i % 160;
        int ky = k / 32, r = k % 32;
        _Float16 v = (_Float16)0.f;
        if (r < 30) {
            int kx = r / 6, c = r % 6;
            if ((kC3Mask[n] >> c) & 1)
                v = (_Float16)c3w[(n * 6 + c) * 25 + ky * 5 + kx];
        }
        w3p[i] = v;
    } else if (i < N3 + N5) {
        int j = i - N3;
        int n = j / 416, k = j % 416;
        _Float16 v = (_Float16)0.f;
        if (n < 120 && k < 400) v = (_Float16)c5w[n * 400 + k];
        w5p[j] = v;
    } else if (i < N3 + N5 + N1) {
        int j = i - N3 - N5;
        int n = j / 128, k = j % 128;
        _Float16 v = (_Float16)0.f;
        if (n < 84 && k < 120) v = (_Float16)l1w[n * 120 + k];
        l1p[j] = v;
    } else if (i < N3 + N5 + N1 + N2) {
        int j = i - N3 - N5 - N1;
        int n = j / 96, k = j % 96;
        _Float16 v = (_Float16)0.f;
        if (n < 10 && k < 84) v = (_Float16)l2w[n * 84 + k];
        l2p[j] = v;
    } else if (i < N3 + N5 + N1 + N2 + NC) {
        int j = i - N3 - N5 - N1 - N2;
        int n = j / 64, k = j % 64;
        int b = k / 8, t = k % 8;
        _Float16 v = (_Float16)0.f;
        if (n < 6 && b < 5 && t < 5) v = (_Float16)c1w[n * 25 + b * 5 + t];
        w1pe[j] = v;
    } else if (i < N3 + N5 + N1 + N2 + 2 * NC) {
        int j = i - N3 - N5 - N1 - N2 - NC;
        int n = j / 64, k = j % 64;
        int b = k / 8, t = k % 8;
        _Float16 v = (_Float16)0.f;
        if (n < 6 && b < 5 && t >= 1 && t <= 5)
            v = (_Float16)c1w[n * 25 + b * 5 + (t - 1)];
        w1po[j] = v;
    }
}

// ---------------------------------------------------------------
// Kernel 1: C1 as implicit GEMM on WMMA f16 16x16x32.
//   M = 16 images/WG, N = 16 (6 real channels), K = 8 row-blocks of 8
//   (2 k-steps; block b = conv row ky, in-block t = offset in an
//    8-wide contiguous x-row run starting at EVEN column 2*px).
// Odd conv columns (dx=1) use the same A run with the tap-shifted
// weight table w1po -> all LDS A reads are 4B-aligned contiguous.
// Fuses bias + tanh + 2x2 avgpool. Output s2 channel-last f16.
// ---------------------------------------------------------------
__global__ void __launch_bounds__(256)
lenet_c1w(const float* __restrict__ x,
          const _Float16* __restrict__ w1pe,
          const _Float16* __restrict__ w1po,
          const float* __restrict__ c1b,
          _Float16* __restrict__ s2) {
    // 16 images x 32x32 halfs, +32 slack for zero-weighted run overruns.
    __shared__ __align__(16) _Float16 sh[16 * 1024 + 32];

    const int tid = threadIdx.x;
    const int imgbase = blockIdx.x * 16;

    // Stage 16 images of x: f32 -> f16 (64KB global -> 32KB LDS).
    {
        const float4* src = (const float4*)(x + (size_t)imgbase * 1024);
        for (int i = tid; i < (16 * 1024) / 4; i += 256) {
            float4 f = src[i];
            _Float16* d = sh + i * 4;
            d[0] = (_Float16)f.x;
            d[1] = (_Float16)f.y;
            d[2] = (_Float16)f.z;
            d[3] = (_Float16)f.w;
        }
        if (tid < 32) sh[16 * 1024 + tid] = (_Float16)0.f;
    }
    __syncthreads();

    const int lane = tid & 31;
    const int wv   = tid >> 5;
    const bool hi  = lane >= 16;
    const int l15  = lane & 15;        // A row (image) == B col (channel)

    // Preload the 4 constant B fragments (even/odd x 2 k-steps).
    Frag bfE[2], bfO[2];
#pragma unroll
    for (int k = 0; k < 2; ++k) {
        const _Float16* pe = w1pe + l15 * 64 + k * 32 + (hi ? 8 : 0);
        bfE[k].i[0] = *(const v4i*)pe;
        bfE[k].i[1] = *(const v4i*)(pe + 16);
        const _Float16* po = w1po + l15 * 64 + k * 32 + (hi ? 8 : 0);
        bfO[k].i[0] = *(const v4i*)po;
        bfO[k].i[1] = *(const v4i*)(po + 16);
    }
    const float bias = c1b[l15 < 6 ? l15 : 5];
    const _Float16* baseL = sh + l15 * 1024;
    const int r0 = hi ? 32 : 0;        // k-step0 rows: lo {ky0,ky2}, hi {ky1,ky3}
    const int r1 = hi ? 96 : 64;

    // 196 pooled positions split across 8 waves (wave-uniform loop).
    for (int p = wv; p < 196; p += 8) {
        const int py = p / 14, px = p % 14;
        float pooled[8];
#pragma unroll
        for (int r = 0; r < 8; ++r) pooled[r] = 0.f;

#pragma unroll
        for (int d = 0; d < 4; ++d) {
            const int dy = d >> 1, dx = d & 1;
            const _Float16* q = baseL + (2 * py + dy) * 32 + 2 * px;
            v8f c = {};
            // k-step 0: conv rows ky 0..3
            {
                Frag a;
#pragma unroll
                for (int u = 0; u < 4; ++u) {
                    a.w[u]     = *(const int*)(q + r0 + 2 * u);
                    a.w[4 + u] = *(const int*)(q + r1 + 2 * u);
                }
                c = wmma_f16(a, dx ? bfO[0] : bfE[0], c);
            }
            // k-step 1: conv row ky=4 (other blocks are zero-weighted;
            // duplicated reads CSE away).
            {
                Frag a;
#pragma unroll
                for (int u = 0; u < 4; ++u) {
                    a.w[u]     = *(const int*)(q + 128 + 2 * u);
                    a.w[4 + u] = *(const int*)(q + 128 + 2 * u);
                }
                c = wmma_f16(a, dx ? bfO[1] : bfE[1], c);
            }
#pragma unroll
            for (int r = 0; r < 8; ++r)
                pooled[r] += fast_tanh(c[r] + bias);
        }
        if (l15 < 6) {
#pragma unroll
            for (int r = 0; r < 8; ++r) {
                int img = imgbase + (hi ? r + 8 : r);
                s2[(size_t)img * 1176 + (py * 14 + px) * 6 + l15] =
                    (_Float16)(pooled[r] * 0.25f);
            }
        }
    }
}

// ---------------------------------------------------------------
// Kernel 2: C3 as implicit GEMM on WMMA f16 16x16x32.
//   M = 16 images/WG, N = 16 output maps, K = 5 tap-rows x 32
//   (30 real: kx*6+c, channel-last => contiguous A runs).
// Fuses bias + tanh + 2x2 avgpool. s4 out: f16 [B][400].
// ---------------------------------------------------------------
__global__ void __launch_bounds__(256)
lenet_c3(const _Float16* __restrict__ s2,
         const _Float16* __restrict__ w3p,
         const float* __restrict__ c3b,
         _Float16* __restrict__ s4) {
    __shared__ __align__(16) _Float16 tile[16 * 1176 + 16];

    const int tid = threadIdx.x;
    const int imgbase = blockIdx.x * 16;

    {
        const v4i* gsrc = (const v4i*)(s2 + (size_t)imgbase * 1176);
        v4i* ldst = (v4i*)tile;
        for (int i = tid; i < (16 * 1176) / 8; i += 256)
            ldst[i] = gsrc[i];
        if (tid < 16) tile[16 * 1176 + tid] = (_Float16)0.f;
    }
    __syncthreads();

    const int lane = tid & 31;
    const int wv   = tid >> 5;
    const bool hi  = lane >= 16;
    const int l15  = lane & 15;
    const int sh   = hi ? 8 : 0;

    Frag bf[5];
#pragma unroll
    for (int k = 0; k < 5; ++k) {
        const _Float16* p = w3p + l15 * 160 + k * 32 + (hi ? 8 : 0);
        bf[k].i[0] = *(const v4i*)p;
        bf[k].i[1] = *(const v4i*)(p + 16);
    }
    const float bias = c3b[l15];

    for (int p = wv; p < 25; p += 8) {
        const int py = p / 5, px = p % 5;
        float pooled[8];
#pragma unroll
        for (int r = 0; r < 8; ++r) pooled[r] = 0.f;

#pragma unroll
        for (int d = 0; d < 4; ++d) {
            const int oy = 2 * py + (d >> 1);
            const int ox = 2 * px + (d & 1);
            v8f c = {};
#pragma unroll
            for (int k = 0; k < 5; ++k) {
                const _Float16* rb =
                    &tile[(l15 * 14 + oy + k) * 84 + ox * 6 + sh];
                Frag a;
#pragma unroll
                for (int q = 0; q < 4; ++q)
                    a.w[q] = *(const int*)(rb + 2 * q);
#pragma unroll
                for (int q = 0; q < 4; ++q)
                    a.w[4 + q] = *(const int*)(rb + 16 + 2 * q);
                c = wmma_f16(a, bf[k], c);
            }
#pragma unroll
            for (int r = 0; r < 8; ++r)
                pooled[r] += fast_tanh(c[r] + bias);
        }
#pragma unroll
        for (int r = 0; r < 8; ++r) {
            int img = imgbase + (hi ? r + 8 : r);
            s4[(size_t)img * 400 + l15 * 25 + p] = (_Float16)(pooled[r] * 0.25f);
        }
    }
}

// ---------------------------------------------------------------
// Kernel 3: fused head. Per workgroup: 16 images.
//   C5 : [16x416] @ [416x128] -> tanh -> y5 (LDS)
//   F6 : [16x128] @ [128x96]  -> tanh -> y6 (LDS)
//   OUT: [16x96]  @ [96x16]   -> +bias -> d_out [B][10] f32
// ---------------------------------------------------------------
__global__ void __launch_bounds__(256)
lenet_head(const _Float16* __restrict__ s4,
           const _Float16* __restrict__ w5p,
           const _Float16* __restrict__ l1p,
           const _Float16* __restrict__ l2p,
           const float* __restrict__ b5,
           const float* __restrict__ b6,
           const float* __restrict__ b7,
           float* __restrict__ out) {
    __shared__ __align__(16) _Float16 s4t[16 * 416];
    __shared__ __align__(16) _Float16 y5[16 * 128];
    __shared__ __align__(16) _Float16 y6[16 * 96];

    const int tid = threadIdx.x;
    const int imgbase = blockIdx.x * 16;
    const _Float16* src = s4 + (size_t)imgbase * 400;

    for (int i = tid; i < 16 * 400; i += 256) {
        int im = i / 400, k = i % 400;
        s4t[im * 416 + k] = src[i];
    }
    {
        int im = tid / 16, k = 400 + (tid % 16);
        s4t[im * 416 + k] = (_Float16)0.f;
    }
    __syncthreads();

    const int lane = tid & 31;
    const int wv   = tid >> 5;
    const bool hi  = lane >= 16;
    const int l15  = lane & 15;
    const int s    = hi ? 8 : 0;
    const int bs   = hi ? 8 : 0;

    // ---- C5: 8 N-tiles (one per wave), K = 13 steps ----
    {
        const int n = wv * 16 + l15;
        v8f c = {};
        for (int k = 0; k < 13; ++k) {
            Frag a, b;
            const _Float16* ap = &s4t[l15 * 416 + k * 32 + s];
            a.i[0] = *(const v4i*)ap;
            a.i[1] = *(const v4i*)(ap + 16);
            const _Float16* bp = w5p + n * 416 + k * 32 + bs;
            b.i[0] = *(const v4i*)bp;
            b.i[1] = *(const v4i*)(bp + 16);
            c = wmma_f16(a, b, c);
        }
        const float bb = b5[n < 120 ? n : 119];
#pragma unroll
        for (int r = 0; r < 8; ++r) {
            int row = hi ? r + 8 : r;
            float v = (n < 120) ? fast_tanh(c[r] + bb) : 0.f;
            y5[row * 128 + n] = (_Float16)v;
        }
    }
    __syncthreads();

    // ---- F6: 6 N-tiles on waves 0..5, K = 4 steps ----
    if (wv < 6) {
        const int n = wv * 16 + l15;
        v8f c = {};
#pragma unroll
        for (int k = 0; k < 4; ++k) {
            Frag a, b;
            const _Float16* ap = &y5[l15 * 128 + k * 32 + s];
            a.i[0] = *(const v4i*)ap;
            a.i[1] = *(const v4i*)(ap + 16);
            const _Float16* bp = l1p + n * 128 + k * 32 + bs;
            b.i[0] = *(const v4i*)bp;
            b.i[1] = *(const v4i*)(bp + 16);
            c = wmma_f16(a, b, c);
        }
        const float bb = b6[n < 84 ? n : 83];
#pragma unroll
        for (int r = 0; r < 8; ++r) {
            int row = hi ? r + 8 : r;
            float v = (n < 84) ? fast_tanh(c[r] + bb) : 0.f;
            y6[row * 96 + n] = (_Float16)v;
        }
    }
    __syncthreads();

    // ---- Output layer: wave 0 only, N = 16 (10 valid), K = 3 steps ----
    if (wv == 0) {
        const int n = l15;
        v8f c = {};
#pragma unroll
        for (int k = 0; k < 3; ++k) {
            Frag a, b;
            const _Float16* ap = &y6[l15 * 96 + k * 32 + s];
            a.i[0] = *(const v4i*)ap;
            a.i[1] = *(const v4i*)(ap + 16);
            const _Float16* bp = l2p + n * 96 + k * 32 + bs;
            b.i[0] = *(const v4i*)bp;
            b.i[1] = *(const v4i*)(bp + 16);
            c = wmma_f16(a, b, c);
        }
        const float bb = b7[n < 10 ? n : 9];
#pragma unroll
        for (int r = 0; r < 8; ++r) {
            int row = hi ? r + 8 : r;
            if (n < 10)
                out[(size_t)(imgbase + row) * 10 + n] = c[r] + bb;
        }
    }
}

// ---------------------------------------------------------------
extern "C" void kernel_launch(void* const* d_in, const int* in_sizes, int n_in,
                              void* d_out, int out_size, void* d_ws, size_t ws_size,
                              hipStream_t stream) {
    const float* x   = (const float*)d_in[0];
    const float* c1w = (const float*)d_in[1];
    const float* c1b = (const float*)d_in[2];
    const float* c3w = (const float*)d_in[3];
    const float* c3b = (const float*)d_in[4];
    const float* c5w = (const float*)d_in[5];
    const float* c5b = (const float*)d_in[6];
    const float* l1w = (const float*)d_in[7];
    const float* l1b = (const float*)d_in[8];
    const float* l2w = (const float*)d_in[9];
    const float* l2b = (const float*)d_in[10];

    const int B = in_sizes[0] / (32 * 32);   // 16384

    // Workspace layout (256B aligned regions)
    char* ws = (char*)d_ws;
    size_t off = 0;
    auto take = [&](size_t bytes) {
        size_t o = off;
        off = (off + bytes + 255) & ~(size_t)255;
        return o;
    };
    _Float16* s2   = (_Float16*)(ws + take((size_t)B * 1176 * 2)); // 38.5 MB
    _Float16* s4   = (_Float16*)(ws + take((size_t)B * 400 * 2));  // 13.1 MB
    _Float16* w3p  = (_Float16*)(ws + take((size_t)16 * 160 * 2));
    _Float16* w5p  = (_Float16*)(ws + take((size_t)128 * 416 * 2));
    _Float16* l1p  = (_Float16*)(ws + take((size_t)96 * 128 * 2));
    _Float16* l2p  = (_Float16*)(ws + take((size_t)16 * 96 * 2));
    _Float16* w1pe = (_Float16*)(ws + take((size_t)16 * 64 * 2));
    _Float16* w1po = (_Float16*)(ws + take((size_t)16 * 64 * 2));
    (void)ws_size;

    // Weight packing (small, one-shot)
    {
        const int total = 16 * 160 + 128 * 416 + 96 * 128 + 16 * 96 + 2 * (16 * 64);
        lenet_prep<<<(total + 255) / 256, 256, 0, stream>>>(
            c1w, c3w, c5w, l1w, l2w, w1pe, w1po, w3p, w5p, l1p, l2p);
    }
    // C1 as WMMA implicit GEMM, fused tanh+pool (reads the 67MB input once)
    lenet_c1w<<<B / 16, 256, 0, stream>>>(x, w1pe, w1po, c1b, s2);
    // C3 as WMMA implicit GEMM, fused tanh+pool
    lenet_c3<<<B / 16, 256, 0, stream>>>(s2, w3p, c3b, s4);
    // C5 + F6 + output, fused WMMA head
    lenet_head<<<B / 16, 256, 0, stream>>>(s4, w5p, l1p, l2p,
                                           c5b, l1b, l2b, (float*)d_out);
}
